// Attention_3599182594919
// MI455X (gfx1250) — matrix operand order repaired and checked by reading, not compiled, not run on hardware
//
#include <hip/hip_runtime.h>
#include <hip/hip_bf16.h>

typedef __bf16 bf16;
typedef __attribute__((ext_vector_type(16))) __bf16 v16bf;
typedef __attribute__((ext_vector_type(8)))  __bf16 v8bf;
typedef __attribute__((ext_vector_type(4)))  __bf16 v4bf;
typedef __attribute__((ext_vector_type(8)))  float  v8f;
typedef __attribute__((ext_vector_type(4)))  float  v4f;
typedef __attribute__((ext_vector_type(4)))  unsigned int u32x4;
typedef __attribute__((ext_vector_type(8)))  int i32x8;
typedef __attribute__((ext_vector_type(4)))  int i32x4;

#define BB 8
#define NN 2048
#define CC 384
#define HH 6
#define DD 64
// SCALE = 64^-0.5 = 0.125f

#if defined(__has_builtin)
#  if __has_builtin(__builtin_amdgcn_tensor_load_to_lds) && \
      __has_builtin(__builtin_amdgcn_s_wait_tensorcnt)
#    define HAVE_TDM 1
#  else
#    define HAVE_TDM 0
#  endif
#else
#  define HAVE_TDM 0
#endif

// ---------------------------------------------------------------------------
// WMMA helpers (CDNA5 16x16x32 bf16, f32 accumulate)
// ---------------------------------------------------------------------------
__device__ __forceinline__ v8f wmma_bf16(v16bf a, v16bf b, v8f c) {
  return __builtin_amdgcn_wmma_f32_16x16x32_bf16(false, a, false, b, (short)0, c,
                                                 false, false);
}

// A-matrix 16x32 bf16 fragment from a row-major row pointer.
// Lane half h holds K in [k0+h*16, k0+h*16+16).
template <typename PT>
__device__ __forceinline__ v16bf load_frag_a(const PT* rowptr, int k0, int half) {
  v8bf lo = *(const v8bf*)(rowptr + k0 + half * 16);
  v8bf hi = *(const v8bf*)(rowptr + k0 + half * 16 + 8);
  v16bf r;
#pragma unroll
  for (int i = 0; i < 8; ++i) { r[i] = lo[i]; r[8 + i] = hi[i]; }
  return r;
}

// B-matrix 32x16 bf16 fragment: the 16 K-values for this lane are contiguous.
template <typename PT>
__device__ __forceinline__ v16bf load_frag_b16(const PT* p) {
  v8bf lo = *(const v8bf*)(p);
  v8bf hi = *(const v8bf*)(p + 8);
  v16bf r;
#pragma unroll
  for (int i = 0; i < 8; ++i) { r[i] = lo[i]; r[8 + i] = hi[i]; }
  return r;
}

// ---------------------------------------------------------------------------
// Tensor Data Mover: 2-D tile Global -> LDS  (D# per CDNA5 ISA ch. 8)
// ---------------------------------------------------------------------------
#if HAVE_TDM
__device__ __forceinline__ void tdm_load_2d(unsigned lds_off, const void* gptr,
                                            unsigned tensor_d0, unsigned tensor_d1,
                                            unsigned tile_d0, unsigned tile_d1,
                                            unsigned stride_elems) {
  unsigned long long ga = (unsigned long long)gptr;
  u32x4 g0;
  g0[0] = 1u;                                   // count=1, user descriptor
  g0[1] = lds_off;                              // lds_addr (bytes)
  g0[2] = (unsigned)(ga & 0xffffffffu);         // global_addr[31:0]
  g0[3] = (unsigned)((ga >> 32) & 0x01ffffffu)  // global_addr[56:32]
          | (2u << 30);                         // type=2 ("image")
  i32x8 g1;
  g1[0] = (int)(1u << 16);                      // data_size=1 (2 bytes)
  g1[1] = (int)(tensor_d0 << 16);               // tensor_dim0[15:0]
  g1[2] = (int)((tensor_d0 >> 16) | (tensor_d1 << 16));
  g1[3] = (int)((tensor_d1 >> 16) | (tile_d0 << 16));
  g1[4] = (int)(tile_d1 & 0xffffu);             // tile_dim1 (tile_dim2=0)
  g1[5] = (int)stride_elems;                    // tensor_dim0_stride[31:0]
  g1[6] = 0;
  g1[7] = 0;
  i32x4 z4 = {0, 0, 0, 0};
#if __clang_major__ >= 23
  i32x8 z8 = {0, 0, 0, 0, 0, 0, 0, 0};
  __builtin_amdgcn_tensor_load_to_lds(g0, g1, z4, z4, z8, 0);
#else
  __builtin_amdgcn_tensor_load_to_lds(g0, g1, z4, z4, 0);
#endif
}
#endif

// ---------------------------------------------------------------------------
// Kernel 0: fp32 -> bf16 conversion (x and both weight matrices)
// ---------------------------------------------------------------------------
__global__ __launch_bounds__(256) void k_cvt_bf16(const float* __restrict__ x,
                                                  bf16* __restrict__ y, int n4) {
  int i = blockIdx.x * blockDim.x + threadIdx.x;
  if (i < n4) {
    v4f v = *(const v4f*)(x + (size_t)i * 4);
    v4bf o;
#pragma unroll
    for (int j = 0; j < 4; ++j) o[j] = (bf16)v[j];
    *(v4bf*)(y + (size_t)i * 4) = o;
  }
}

// ---------------------------------------------------------------------------
// Kernel 1: QKV GEMM: x_bf(16384x384) @ qkv_w_bf^T(384x1152).
// Wave tile 32x64 (2 A-frags, 8 accumulators -> 8 WMMA per K-step, B frags
// reused 2x).  64-col tiles never straddle q/k/v (384-aligned) or head
// (64-aligned) boundaries, so `which` and `h` are block-uniform.
// ---------------------------------------------------------------------------
__global__ __launch_bounds__(256) void k_qkv_gemm(const bf16* __restrict__ xb,
                                                  const bf16* __restrict__ wb,
                                                  bf16* __restrict__ q,
                                                  bf16* __restrict__ kmat,
                                                  bf16* __restrict__ vt) {
  const int wv = threadIdx.x >> 5, lane = threadIdx.x & 31;
  const int half = lane >> 4, l16 = lane & 15;
  const int m0 = blockIdx.y * 256 + wv * 32;  // 32 rows per wave
  const int j0 = blockIdx.x * 64;             // 64 cols

  v8f acc[2][4] = {};
  const bf16* arow0 = xb + (size_t)(m0 + l16) * CC;
  const bf16* arow1 = xb + (size_t)(m0 + 16 + l16) * CC;
  const bf16* brow  = wb + (size_t)(j0 + l16) * CC + half * 16;

  for (int kk = 0; kk < CC; kk += 32) {
    if (kk + 32 < CC) {
      __builtin_prefetch(arow0 + kk + 32, 0, 3);
      __builtin_prefetch(arow1 + kk + 32, 0, 3);
    }
    v16bf af0 = load_frag_a(arow0, kk, half);
    v16bf af1 = load_frag_a(arow1, kk, half);
    v16bf bf4[4];
#pragma unroll
    for (int c = 0; c < 4; ++c)
      bf4[c] = load_frag_b16(brow + (size_t)c * 16 * CC + kk);
#pragma unroll
    for (int c = 0; c < 4; ++c) {
      acc[0][c] = wmma_bf16(af0, bf4[c], acc[0][c]);
      acc[1][c] = wmma_bf16(af1, bf4[c], acc[1][c]);
    }
  }

  // Block-uniform destination selection.
  const int which = j0 / CC;        // 0=q, 1=k, 2=v
  const int h = (j0 % CC) >> 6;     // head (64-aligned tile)
  bf16* dst = (which == 0) ? q : (which == 1) ? kmat : vt;

#pragma unroll
  for (int g = 0; g < 2; ++g) {
#pragma unroll
    for (int c = 0; c < 4; ++c) {
      const int d = c * 16 + l16;
#pragma unroll
      for (int r = 0; r < 8; ++r) {
        const int m = m0 + g * 16 + r + 8 * half;
        const int b = m >> 11, n = m & (NN - 1);
        const size_t head = (size_t)b * HH + h;
        const bf16 val = (bf16)acc[g][c][r];
        if (which < 2) dst[(head * NN + n) * DD + d] = val;
        else           dst[(head * DD + d) * NN + n] = val;  // V transposed
      }
    }
  }
}

// ---------------------------------------------------------------------------
// Kernel 2: fused flash attention.  8 waves x 16 query rows per block.
// K/V tiles staged into LDS by the Tensor Data Mover (double-buffered,
// TENSORcnt-fenced), fragments read via ds_load_b128.  Online softmax with
// -1e5*mask bias (non-temporal mask loads).
// ---------------------------------------------------------------------------
__global__ __launch_bounds__(256) void k_attn(const bf16* __restrict__ q,
                                              const bf16* __restrict__ kmat,
                                              const bf16* __restrict__ vt,
                                              const float* __restrict__ mask,
                                              bf16* __restrict__ ao) {
  __shared__ __align__(16) bf16 kstage[2][32 * 64];  // [key][d]   4KB x2
  __shared__ __align__(16) bf16 vstage[2][64 * 32];  // [d][key]   4KB x2
  __shared__ __align__(16) bf16 plds[8][16 * 32];    // P staging  8KB

  const int tid = threadIdx.x;
  const int wv = tid >> 5, lane = tid & 31;
  const int half = lane >> 4, l16 = lane & 15;
  const int bh = blockIdx.y;
  const int b = bh / HH, h = bh % HH;
  const int q0 = blockIdx.x * 128 + wv * 16;

  const bf16* qp = q    + (size_t)bh * NN * DD;
  const bf16* kp = kmat + (size_t)bh * NN * DD;
  const bf16* vp = vt   + (size_t)bh * DD * NN;
  const float* mp = mask + (size_t)b * NN * NN;

  auto stage = [&](int buf, int t) {
#if HAVE_TDM
    if (wv == 0) {
      tdm_load_2d((unsigned)(size_t)&kstage[buf][0], kp + (size_t)t * DD,
                  /*tensor_d0=*/DD, /*tensor_d1=*/NN,
                  /*tile_d0=*/DD, /*tile_d1=*/32, /*stride=*/DD);
      tdm_load_2d((unsigned)(size_t)&vstage[buf][0], vp + t,
                  /*tensor_d0=*/NN, /*tensor_d1=*/DD,
                  /*tile_d0=*/32, /*tile_d1=*/DD, /*stride=*/NN);
    }
#else
    *(v8bf*)(&kstage[buf][0] + tid * 8) =
        *(const v8bf*)(kp + (size_t)t * DD + tid * 8);
    {
      const int row = tid >> 2, chunk = tid & 3;
      *(v8bf*)(&vstage[buf][0] + row * 32 + chunk * 8) =
          *(const v8bf*)(vp + (size_t)row * NN + t + chunk * 8);
    }
#endif
  };

  const bf16* qrow = qp + (size_t)(q0 + l16) * DD;
  const v16bf qa0 = load_frag_a(qrow, 0, half);
  const v16bf qa1 = load_frag_a(qrow, 32, half);

  v8f o[4] = {};
  float Mrow[8], Lrow[8];
#pragma unroll
  for (int r = 0; r < 8; ++r) { Mrow[r] = -3.0e38f; Lrow[r] = 0.0f; }

  stage(0, 0);
#if HAVE_TDM
  if (wv == 0) __builtin_amdgcn_s_wait_tensorcnt(0);
#endif
  __syncthreads();

  for (int t = 0; t < NN; t += 32) {
    const int cur = (t >> 5) & 1;
    if (t + 32 < NN) stage(cur ^ 1, t + 32);  // TDM prefetch next tile

    const bf16* kb = &kstage[cur][0];
    const bf16* vb = &vstage[cur][0];

    // ---- scores: S = Q @ K^T (two 16x16 C tiles over 32 keys) ----
    v8f s[2];
#pragma unroll
    for (int nc = 0; nc < 2; ++nc) {
      const bf16* krow = kb + (nc * 16 + l16) * DD;
      v16bf kb0 = load_frag_b16(krow + half * 16);
      v16bf kb1 = load_frag_b16(krow + 32 + half * 16);
      v8f ss = {};
      ss = wmma_bf16(qa0, kb0, ss);
      ss = wmma_bf16(qa1, kb1, ss);
      s[nc] = ss;
    }

    // ---- scale + mask bias (streamed once -> non-temporal loads) ----
#pragma unroll
    for (int r = 0; r < 8; ++r) {
      const size_t qr = (size_t)(q0 + r + 8 * half) * NN + t;
      const float m0v = __builtin_nontemporal_load(mp + qr + l16);
      const float m1v = __builtin_nontemporal_load(mp + qr + 16 + l16);
      s[0][r] = s[0][r] * 0.125f - 100000.0f * m0v;
      s[1][r] = s[1][r] * 0.125f - 100000.0f * m1v;
    }

    // ---- online softmax ----
#pragma unroll
    for (int r = 0; r < 8; ++r) {
      float mx = fmaxf(s[0][r], s[1][r]);
#pragma unroll
      for (int off = 1; off < 16; off <<= 1)
        mx = fmaxf(mx, __shfl_xor(mx, off, 32));
      const float newM = fmaxf(Mrow[r], mx);
      const float alpha = __expf(Mrow[r] - newM);
      const float p0 = __expf(s[0][r] - newM);
      const float p1 = __expf(s[1][r] - newM);
      s[0][r] = p0; s[1][r] = p1;
      float rs = p0 + p1;
#pragma unroll
      for (int off = 1; off < 16; off <<= 1)
        rs += __shfl_xor(rs, off, 32);
      Lrow[r] = Lrow[r] * alpha + rs;
      Mrow[r] = newM;
#pragma unroll
      for (int c = 0; c < 4; ++c) o[c][r] *= alpha;
    }

    // ---- P: C layout -> A layout via per-wave LDS tile ----
    bf16* pb = &plds[wv][0];
#pragma unroll
    for (int r = 0; r < 8; ++r) {
      pb[(r + 8 * half) * 32 + l16]      = (bf16)s[0][r];
      pb[(r + 8 * half) * 32 + 16 + l16] = (bf16)s[1][r];
    }
    asm volatile("s_wait_dscnt 0" ::: "memory");
    const v16bf pa = load_frag_a(pb + l16 * 32, 0, half);

    // ---- O += P @ V ----
#pragma unroll
    for (int c = 0; c < 4; ++c) {
      v16bf vf = load_frag_b16(vb + (c * 16 + l16) * 32 + half * 16);
      o[c] = wmma_bf16(pa, vf, o[c]);
    }

#if HAVE_TDM
    if (wv == 0) __builtin_amdgcn_s_wait_tensorcnt(0);
#endif
    __syncthreads();
  }

  // ---- normalize and store to ao[b][n][h*64+d] ----
#pragma unroll
  for (int c = 0; c < 4; ++c) {
#pragma unroll
    for (int r = 0; r < 8; ++r) {
      const int n = q0 + r + 8 * half;
      const int d = c * 16 + l16;
      ao[((size_t)b * NN + n) * CC + h * DD + d] = (bf16)(o[c][r] / Lrow[r]);
    }
  }
}

// ---------------------------------------------------------------------------
// Kernel 3: projection: ao(16384x384) @ proj_w_bf^T + bias -> fp32 out (NT)
// Wave tile 32x64 like the QKV GEMM.
// ---------------------------------------------------------------------------
__global__ __launch_bounds__(256) void k_proj_gemm(const bf16* __restrict__ ao,
                                                   const bf16* __restrict__ wb,
                                                   const float* __restrict__ bias,
                                                   float* __restrict__ out) {
  const int wv = threadIdx.x >> 5, lane = threadIdx.x & 31;
  const int half = lane >> 4, l16 = lane & 15;
  const int m0 = blockIdx.y * 256 + wv * 32;
  const int j0 = blockIdx.x * 64;

  v8f acc[2][4] = {};
  const bf16* arow0 = ao + (size_t)(m0 + l16) * CC;
  const bf16* arow1 = ao + (size_t)(m0 + 16 + l16) * CC;
  const bf16* brow  = wb + (size_t)(j0 + l16) * CC + half * 16;

  for (int kk = 0; kk < CC; kk += 32) {
    if (kk + 32 < CC) {
      __builtin_prefetch(arow0 + kk + 32, 0, 3);
      __builtin_prefetch(arow1 + kk + 32, 0, 3);
    }
    v16bf af0 = load_frag_a(arow0, kk, half);
    v16bf af1 = load_frag_a(arow1, kk, half);
    v16bf bf4[4];
#pragma unroll
    for (int c = 0; c < 4; ++c)
      bf4[c] = load_frag_b16(brow + (size_t)c * 16 * CC + kk);
#pragma unroll
    for (int c = 0; c < 4; ++c) {
      acc[0][c] = wmma_bf16(af0, bf4[c], acc[0][c]);
      acc[1][c] = wmma_bf16(af1, bf4[c], acc[1][c]);
    }
  }

#pragma unroll
  for (int c = 0; c < 4; ++c) {
    const int j = j0 + c * 16 + l16;
    const float bj = bias[j];
#pragma unroll
    for (int g = 0; g < 2; ++g) {
#pragma unroll
      for (int r = 0; r < 8; ++r) {
        const int m = m0 + g * 16 + r + 8 * half;
        __builtin_nontemporal_store(acc[g][c][r] + bj, out + (size_t)m * CC + j);
      }
    }
  }
}

// ---------------------------------------------------------------------------
// Launch
// ---------------------------------------------------------------------------
extern "C" void kernel_launch(void* const* d_in, const int* in_sizes, int n_in,
                              void* d_out, int out_size, void* d_ws, size_t ws_size,
                              hipStream_t stream) {
  const float* x      = (const float*)d_in[0];
  // d_in[1] = prompt (unused)
  const float* mask   = (const float*)d_in[2];
  const float* qkv_w  = (const float*)d_in[3];
  const float* proj_w = (const float*)d_in[4];
  const float* proj_b = (const float*)d_in[5];
  float* out = (float*)d_out;

  const size_t NE  = (size_t)BB * NN * CC;  // 6,291,456 (== B*H*N*D)
  const size_t NWQ = (size_t)3 * CC * CC;   // 442,368
  const size_t NWP = (size_t)CC * CC;       // 147,456
  bf16* xb  = (bf16*)d_ws;
  bf16* qb  = xb + NE;
  bf16* kb  = qb + NE;
  bf16* vt  = kb + NE;
  bf16* ao  = vt + NE;
  bf16* wqb = ao + NE;
  bf16* wpb = wqb + NWQ;

  k_cvt_bf16<<<(int)(NE / 4 + 255) / 256, 256, 0, stream>>>(x, xb, (int)(NE / 4));
  k_cvt_bf16<<<(int)(NWQ / 4 + 255) / 256, 256, 0, stream>>>(qkv_w, wqb, (int)(NWQ / 4));
  k_cvt_bf16<<<(int)(NWP / 4 + 255) / 256, 256, 0, stream>>>(proj_w, wpb, (int)(NWP / 4));

  // 1) QKV GEMM (M=16384, N=1152, K=384), 256-row blocks
  {
    dim3 grid(1152 / 64, (BB * NN) / 256);
    k_qkv_gemm<<<grid, 256, 0, stream>>>(xb, wqb, qb, kb, vt);
  }
  // 2) fused attention (TDM-staged K/V tiles)
  {
    dim3 grid(NN / 128, BB * HH);
    k_attn<<<grid, 256, 0, stream>>>(qb, kb, vt, mask, ao);
  }
  // 3) projection (M=16384, N=384, K=384), 256-row blocks
  {
    dim3 grid(CC / 64, (BB * NN) / 256);
    k_proj_gemm<<<grid, 256, 0, stream>>>(ao, wpb, proj_b, out);
  }
}
